// DifferentialTransformer_69114613727322
// MI455X (gfx1250) — matrix operand order repaired and checked by reading, not compiled, not run on hardware
//
#include <hip/hip_runtime.h>
#include <cmath>

// ---------------------------------------------------------------------------
// Types for CDNA5 WMMA (wave32, 16x16x32 bf16 -> f32)
// ---------------------------------------------------------------------------
typedef __bf16 bf16;
typedef __attribute__((ext_vector_type(16))) __bf16 v16bf;
typedef __attribute__((ext_vector_type(8)))  __bf16 v8bf;
typedef __attribute__((ext_vector_type(8)))  float  v8f;
typedef int v4i_vs __attribute__((__vector_size__(16)));  // matches builtin sig

#define DEV static __device__ __forceinline__

// ---------------------------------------------------------------------------
// CDNA5 async global->LDS copy path (ASYNCcnt-tracked DMA).
// Builtin signature (from compiler diagnostic): (int4 __device__*, int4
// __shared__*, imm offset, imm cpol). Generic pointers convert implicitly to
// the CUDA language address spaces in HIP, so plain v4i_vs* casts suffice.
// ---------------------------------------------------------------------------
#if defined(__has_builtin)
#if __has_builtin(__builtin_amdgcn_global_load_async_to_lds_b128) && \
    __has_builtin(__builtin_amdgcn_s_wait_asynccnt)
#define HAVE_ASYNC_LDS 1
#endif
#endif

#ifdef HAVE_ASYNC_LDS
// copy 16B per call; imm offset applies to BOTH global and LDS address
#define ASYNC_CP16(gsrc, ldst, off)                                          \
  __builtin_amdgcn_global_load_async_to_lds_b128(                            \
      (v4i_vs*)(void*)(gsrc), (v4i_vs*)(void*)(ldst), (off), 0)
#define ASYNC_WAIT() __builtin_amdgcn_s_wait_asynccnt(0)
#endif

DEV v16bf ld_frag(const bf16* lo, const bf16* hi) {
  v8bf a = *(const v8bf*)lo;
  v8bf b = *(const v8bf*)hi;
  v16bf r;
#pragma unroll
  for (int i = 0; i < 8; ++i) { r[i] = a[i]; r[i + 8] = b[i]; }
  return r;
}

DEV v8f wmma_bf16(v16bf a, v16bf b, v8f c) {
  // D = A(16x32) * B(32x16) + C, f32 accumulate
  return __builtin_amdgcn_wmma_f32_16x16x32_bf16(false, a, false, b,
                                                 (short)0, c, false, false);
}

// ---------------------------------------------------------------------------
// Generic GEMM: C[M,N](f32 or bf16) = A[M,K](bf16, row-major)
//                                   x W[K,N](f32, row-major, cvt->bf16 in LDS)
// Tile 128x128x32, 256 threads = 8 waves, each wave computes 32(M)x64(N).
// EPI: 0=none, 1=+aux[row,col] residual, 2=+aux[col] bias.
// grid.z batches A/W/C by element strides sA/sW/sC (per-head QKV).
// ---------------------------------------------------------------------------
template <int OUT_BF16, int EPI>
__global__ void __launch_bounds__(256)
gemm_kernel(const bf16* __restrict__ A, const float* __restrict__ W,
            void* __restrict__ Cout, const float* __restrict__ aux,
            int M, int N, int K, long sA, long sW, long sC) {
  __shared__ bf16 As[128][40];   // [m][k], pad 32->40
  __shared__ bf16 Bs[128][40];   // [n][k] transposed, pad 32->40

  const int tid = threadIdx.x;
  const int z = blockIdx.z;
  A += (long)z * sA;
  W += (long)z * sW;
  bf16*  Cb = (bf16*)Cout + (long)z * sC;
  float* Cf = (float*)Cout + (long)z * sC;

  const int n0 = blockIdx.x * 128;
  const int m0 = blockIdx.y * 128;
  const int wave = tid >> 5, lane = tid & 31;
  const int wm = wave >> 1, wn = wave & 1;  // 4x2 wave grid
  const int lh = lane >> 4, ll = lane & 15;

  v8f acc[2][4];
#pragma unroll
  for (int i = 0; i < 2; ++i)
#pragma unroll
    for (int j = 0; j < 4; ++j)
#pragma unroll
      for (int e = 0; e < 8; ++e) acc[i][j][e] = 0.f;

  const int arow = tid >> 1, akoff = (tid & 1) * 16;  // A staging
  const int bkk = tid >> 3, bnc = tid & 7;            // B staging

  for (int k0 = 0; k0 < K; k0 += 32) {
    // --- stage A tile (bf16): async DMA to LDS when available ---
    const bf16* ap = A + (long)(m0 + arow) * K + k0 + akoff;
#ifdef HAVE_ASYNC_LDS
    ASYNC_CP16(ap, &As[arow][akoff], 0);
    ASYNC_CP16(ap, &As[arow][akoff], 16);
#else
    *(v8bf*)&As[arow][akoff]     = *(const v8bf*)ap;
    *(v8bf*)&As[arow][akoff + 8] = *(const v8bf*)(ap + 8);
#endif
    // --- stage W tile, f32 -> bf16, transposed scatter (VALU path) ---
    const float* wp = W + (long)(k0 + bkk) * N + n0 + bnc * 16;
#pragma unroll
    for (int i = 0; i < 16; ++i) Bs[bnc * 16 + i][bkk] = (bf16)wp[i];

    if (k0 + 32 < K) {  // gfx1250 global_prefetch_b8 for next tiles
      __builtin_prefetch(ap + 32, 0, 3);
      __builtin_prefetch(wp + 32L * N, 0, 3);
    }
#ifdef HAVE_ASYNC_LDS
    ASYNC_WAIT();
#endif
    __syncthreads();

    v16bf af[2], bfr[4];
#pragma unroll
    for (int mi = 0; mi < 2; ++mi) {
      const bf16* p = &As[wm * 32 + mi * 16 + ll][lh * 16];
      af[mi] = ld_frag(p, p + 8);  // K: lh*16+0..15 contiguous
    }
#pragma unroll
    for (int ni = 0; ni < 4; ++ni) {
      const bf16* p = &Bs[wn * 64 + ni * 16 + ll][lh * 16];
      bfr[ni] = ld_frag(p, p + 8);  // K: lh*16+0..15 contiguous
    }
#pragma unroll
    for (int mi = 0; mi < 2; ++mi)
#pragma unroll
      for (int ni = 0; ni < 4; ++ni)
        acc[mi][ni] = wmma_bf16(af[mi], bfr[ni], acc[mi][ni]);
    __syncthreads();
  }

#pragma unroll
  for (int mi = 0; mi < 2; ++mi)
#pragma unroll
    for (int ni = 0; ni < 4; ++ni) {
      const int row = m0 + wm * 32 + mi * 16 + lh * 8;
      const int col = n0 + wn * 64 + ni * 16 + ll;
#pragma unroll
      for (int r = 0; r < 8; ++r) {
        float v = acc[mi][ni][r];
        const long idx = (long)(row + r) * N + col;
        if (EPI == 1) v += aux[idx];
        if (EPI == 2) v += aux[col];
        if (OUT_BF16) Cb[idx] = (bf16)v;
        else          Cf[idx] = v;
      }
    }
}

// ---------------------------------------------------------------------------
// Flash attention (one softmax component): O_c = softmax(Q_c K_c^T * s) @ V
// Q/K/V layout: [H][B*S][256] bf16; component c uses dims c*128..c*128+127
// of Q/K, full 256 dims of V. Block = 8 waves, each wave owns 16 query rows.
// K/V key-tiles (32 keys) staged cooperatively in LDS.
// ---------------------------------------------------------------------------
__global__ void __launch_bounds__(256)
flash_kernel(const bf16* __restrict__ Q, const bf16* __restrict__ Kb,
             const bf16* __restrict__ Vb, float* __restrict__ O1,
             float* __restrict__ O2, float scale, int S, int BS) {
  __shared__ bf16 Ks[32][136];      // [key][dim(component)] pad 128->136
  __shared__ bf16 Vt[256][40];      // [dim][key] pad 32->40
  __shared__ bf16 Pb[8][16][32];    // per-wave P transpose bounce

  const int tid = threadIdx.x;
  const int wave = tid >> 5, lane = tid & 31;
  const int lh = lane >> 4, ll = lane & 15;
  const int h = blockIdx.y;
  const int b = blockIdx.z >> 1, c = blockIdx.z & 1;
  const long base = (long)h * BS * 256 + (long)b * S * 256;
  const int qr0 = blockIdx.x * 128 + wave * 16;

  // Preload Q fragments (A-matrix, 16 rows x 128 dims -> 4 frags)
  v16bf qf[4];
  {
    const bf16* qp = Q + base + (long)(qr0 + ll) * 256 + c * 128;
#pragma unroll
    for (int kk = 0; kk < 4; ++kk) {
      const bf16* p = qp + kk * 32 + lh * 16;
      qf[kk] = ld_frag(p, p + 8);
    }
  }

  float Mr[8], L[8];
#pragma unroll
  for (int r = 0; r < 8; ++r) { Mr[r] = -3.0e38f; L[r] = 0.f; }
  v8f Oa[16];
#pragma unroll
  for (int f = 0; f < 16; ++f)
#pragma unroll
    for (int r = 0; r < 8; ++r) Oa[f][r] = 0.f;

  const int skey = tid >> 3, sdc = tid & 7;
  for (int j = 0; j < S; j += 32) {
    __syncthreads();
    {  // stage K tile [32 keys][128 dims of component c] (async DMA if avail)
      const bf16* kp = Kb + base + (long)(j + skey) * 256 + c * 128 + sdc * 16;
#ifdef HAVE_ASYNC_LDS
      ASYNC_CP16(kp, &Ks[skey][sdc * 16], 0);
      ASYNC_CP16(kp, &Ks[skey][sdc * 16], 16);
#else
      *(v8bf*)&Ks[skey][sdc * 16]     = *(const v8bf*)kp;
      *(v8bf*)&Ks[skey][sdc * 16 + 8] = *(const v8bf*)(kp + 8);
#endif
    }
    {  // stage V tile transposed: Vt[dim][key] (needs transpose -> VALU path)
      const bf16* vp = Vb + base + (long)(j + skey) * 256 + sdc * 32;
      v8bf t0 = *(const v8bf*)vp;
      v8bf t1 = *(const v8bf*)(vp + 8);
      v8bf t2 = *(const v8bf*)(vp + 16);
      v8bf t3 = *(const v8bf*)(vp + 24);
#pragma unroll
      for (int i = 0; i < 8; ++i) {
        Vt[sdc * 32 + i][skey]      = t0[i];
        Vt[sdc * 32 + i + 8][skey]  = t1[i];
        Vt[sdc * 32 + i + 16][skey] = t2[i];
        Vt[sdc * 32 + i + 24][skey] = t3[i];
      }
    }
#ifdef HAVE_ASYNC_LDS
    ASYNC_WAIT();
#endif
    __syncthreads();

    // S-tiles: 16 queries x 32 keys
    v8f s0, s1;
#pragma unroll
    for (int r = 0; r < 8; ++r) { s0[r] = 0.f; s1[r] = 0.f; }
#pragma unroll
    for (int kk = 0; kk < 4; ++kk) {
      const bf16* p0 = &Ks[ll][kk * 32 + lh * 16];
      const bf16* p1 = &Ks[16 + ll][kk * 32 + lh * 16];
      s0 = wmma_bf16(qf[kk], ld_frag(p0, p0 + 8), s0);
      s1 = wmma_bf16(qf[kk], ld_frag(p1, p1 + 8), s1);
    }

    // Online softmax (row = lh*8 + r, cols spread over 16 lanes)
    float alpha[8];
#pragma unroll
    for (int r = 0; r < 8; ++r) {
      float a0 = s0[r] * scale, a1 = s1[r] * scale;
      float mx = fmaxf(a0, a1);
#pragma unroll
      for (int m = 1; m < 16; m <<= 1) mx = fmaxf(mx, __shfl_xor(mx, m, 32));
      const float nM = fmaxf(Mr[r], mx);
      alpha[r] = __expf(Mr[r] - nM);
      const float p0 = __expf(a0 - nM), p1 = __expf(a1 - nM);
      float ps = p0 + p1;
#pragma unroll
      for (int m = 1; m < 16; m <<= 1) ps += __shfl_xor(ps, m, 32);
      L[r] = L[r] * alpha[r] + ps;
      Mr[r] = nM;
      Pb[wave][lh * 8 + r][ll]      = (bf16)p0;  // C-layout -> LDS
      Pb[wave][lh * 8 + r][16 + ll] = (bf16)p1;
    }
#pragma unroll
    for (int f = 0; f < 16; ++f)
#pragma unroll
      for (int r = 0; r < 8; ++r) Oa[f][r] *= alpha[r];

    // Read P back as A-fragment (DS ops in-order within the wave)
    const bf16* pp = &Pb[wave][ll][lh * 16];
    const v16bf pf = ld_frag(pp, pp + 8);

    // O += P(16x32) @ V(32x256)
#pragma unroll
    for (int f = 0; f < 16; ++f) {
      const bf16* vp = &Vt[f * 16 + ll][lh * 16];
      Oa[f] = wmma_bf16(pf, ld_frag(vp, vp + 8), Oa[f]);
    }
  }

  float inv[8];
#pragma unroll
  for (int r = 0; r < 8; ++r) inv[r] = 1.f / L[r];
  float* Op = (c ? O2 : O1) + base;
#pragma unroll
  for (int f = 0; f < 16; ++f)
#pragma unroll
    for (int r = 0; r < 8; ++r)
      Op[(long)(qr0 + lh * 8 + r) * 256 + f * 16 + ll] = Oa[f][r] * inv[r];
}

// ---------------------------------------------------------------------------
// Combine: o = o1 - lam[h]*o2, per-head RMS over 256 dims, * (1-lambda_init),
// pack to [BS, H*256] bf16 for the Wo GEMM.
// ---------------------------------------------------------------------------
__global__ void __launch_bounds__(256)
combine_kernel(const float* __restrict__ O1, const float* __restrict__ O2,
               const float* __restrict__ lam, bf16* __restrict__ out,
               int BS, float outScale) {
  __shared__ float red[8];
  const int row = blockIdx.x;
  const int tid = threadIdx.x;
  for (int h = 0; h < 12; ++h) {
    const long idx = (long)h * BS * 256 + (long)row * 256 + tid;
    const float v = O1[idx] - lam[h] * O2[idx];
    float ss = v * v;
#pragma unroll
    for (int m = 1; m < 32; m <<= 1) ss += __shfl_xor(ss, m, 32);
    if ((tid & 31) == 0) red[tid >> 5] = ss;
    __syncthreads();
    float tot = 0.f;
#pragma unroll
    for (int i = 0; i < 8; ++i) tot += red[i];
    const float rstd = rsqrtf(tot * (1.f / 256.f) + 1e-6f);
    out[(long)row * 3072 + h * 256 + tid] = (bf16)(v * rstd * outScale);
    __syncthreads();
  }
}

// ---------------------------------------------------------------------------
// Weightless RMSNorm (f32 in -> bf16 out), one block per row, D = 1536.
// ---------------------------------------------------------------------------
__global__ void __launch_bounds__(256)
rmsnorm_kernel(const float* __restrict__ x, bf16* __restrict__ out, int D) {
  __shared__ float red[8];
  const int row = blockIdx.x, tid = threadIdx.x;
  const float* xr = x + (long)row * D;
  const int nPer = D / 256;
  float v[8];
  float ss = 0.f;
  for (int i = 0; i < nPer; ++i) { v[i] = xr[tid + i * 256]; ss += v[i] * v[i]; }
#pragma unroll
  for (int m = 1; m < 32; m <<= 1) ss += __shfl_xor(ss, m, 32);
  if ((tid & 31) == 0) red[tid >> 5] = ss;
  __syncthreads();
  float tot = 0.f;
#pragma unroll
  for (int i = 0; i < 8; ++i) tot += red[i];
  const float rstd = rsqrtf(tot / (float)D + 1e-6f);
  for (int i = 0; i < nPer; ++i)
    out[(long)row * D + tid + i * 256] = (bf16)(v[i] * rstd);
}

// lam[h] = exp(lq1.lk1) + exp(lq2.lk2) + lambda_init   (dot over HD=128)
__global__ void __launch_bounds__(128)
lambda_kernel(const float* __restrict__ lq1, const float* __restrict__ lq2,
              const float* __restrict__ lk1, const float* __restrict__ lk2,
              float* __restrict__ lam, float lamInit) {
  __shared__ float red[2][4];
  const int h = blockIdx.x, d = threadIdx.x;
  float s1 = lq1[h * 128 + d] * lk1[h * 128 + d];
  float s2 = lq2[h * 128 + d] * lk2[h * 128 + d];
#pragma unroll
  for (int m = 1; m < 32; m <<= 1) {
    s1 += __shfl_xor(s1, m, 32);
    s2 += __shfl_xor(s2, m, 32);
  }
  if ((d & 31) == 0) { red[0][d >> 5] = s1; red[1][d >> 5] = s2; }
  __syncthreads();
  if (d == 0) {
    float a = 0.f, b = 0.f;
#pragma unroll
    for (int i = 0; i < 4; ++i) { a += red[0][i]; b += red[1][i]; }
    lam[h] = __expf(a) + __expf(b) + lamInit;
  }
}

__global__ void swiglu_kernel(const float* __restrict__ g1,
                              const float* __restrict__ g3,
                              bf16* __restrict__ u, long n) {
  const long i = (long)blockIdx.x * blockDim.x + threadIdx.x;
  if (i < n) {
    const float a = g1[i];
    u[i] = (bf16)((a / (1.f + __expf(-a))) * g3[i]);
  }
}

__global__ void f32_to_bf16_kernel(const float* __restrict__ x,
                                   bf16* __restrict__ y, long n) {
  const long i = (long)blockIdx.x * blockDim.x + threadIdx.x;
  if (i < n) y[i] = (bf16)x[i];
}

// ---------------------------------------------------------------------------
// Host orchestration
// ---------------------------------------------------------------------------
extern "C" void kernel_launch(void* const* d_in, const int* in_sizes, int n_in,
                              void* d_out, int out_size, void* d_ws,
                              size_t ws_size, hipStream_t stream) {
  (void)in_sizes; (void)n_in; (void)out_size; (void)ws_size;
  constexpr int Bc = 2, Sc = 2048, Dc = 1536, Hc = 12, HDc = 128,
                HFFc = 4096, Vc = 32000;
  constexpr int BS = Bc * Sc;  // 4096
  const float LAMBDA_INIT = 0.8f - 0.6f * expf(-0.3f * 2.f);
  const float scale = 1.f / sqrtf((float)HDc);

  const float* x    = (const float*)d_in[0];
  const float* Wq   = (const float*)d_in[1];
  const float* Wk   = (const float*)d_in[2];
  const float* Wv   = (const float*)d_in[3];
  const float* lq1  = (const float*)d_in[4];
  const float* lq2  = (const float*)d_in[5];
  const float* lk1  = (const float*)d_in[6];
  const float* lk2  = (const float*)d_in[7];
  const float* Wo   = (const float*)d_in[8];
  const float* w1   = (const float*)d_in[9];
  const float* w2   = (const float*)d_in[10];
  const float* w3   = (const float*)d_in[11];
  const float* Wout = (const float*)d_in[12];
  const float* bout = (const float*)d_in[13];
  float* out = (float*)d_out;

  char* wsp = (char*)d_ws;
  auto alloc = [&](size_t bytes) -> void* {
    void* p = (void*)wsp;
    wsp += (bytes + 255) & ~(size_t)255;
    return p;
  };
  bf16*  xb  = (bf16*)alloc((size_t)BS * Dc * 2);
  bf16*  hb  = (bf16*)alloc((size_t)BS * Dc * 2);
  bf16*  qb  = (bf16*)alloc((size_t)Hc * BS * 256 * 2);
  bf16*  kb  = (bf16*)alloc((size_t)Hc * BS * 256 * 2);
  bf16*  vb  = (bf16*)alloc((size_t)Hc * BS * 256 * 2);
  float* o1  = (float*)alloc((size_t)Hc * BS * 256 * 4);
  float* o2  = (float*)alloc((size_t)Hc * BS * 256 * 4);
  bf16*  ob  = (bf16*)alloc((size_t)BS * 2 * Dc * 2);
  float* yb  = (float*)alloc((size_t)BS * Dc * 4);
  float* g1  = (float*)alloc((size_t)BS * HFFc * 4);
  float* g3  = (float*)alloc((size_t)BS * HFFc * 4);
  bf16*  ub  = (bf16*)alloc((size_t)BS * HFFc * 2);
  float* xA  = (float*)alloc((size_t)BS * Dc * 4);
  float* xB  = (float*)alloc((size_t)BS * Dc * 4);
  float* lam = (float*)alloc(256);

  const dim3 blk(256);
  const float* xcur = x;
  float* xnext[2] = {xA, xB};

  for (int l = 0; l < 2; ++l) {
    rmsnorm_kernel<<<BS, blk, 0, stream>>>(xcur, xb, Dc);
    lambda_kernel<<<Hc, 128, 0, stream>>>(
        lq1 + (long)l * Hc * HDc, lq2 + (long)l * Hc * HDc,
        lk1 + (long)l * Hc * HDc, lk2 + (long)l * Hc * HDc, lam, LAMBDA_INIT);

    // QKV projections, batched over heads via grid.z
    const long woff = (long)l * Hc * Dc * 2 * HDc;
    const dim3 gq(2 * HDc / 128, BS / 128, Hc);
    gemm_kernel<1, 0><<<gq, blk, 0, stream>>>(
        xb, Wq + woff, qb, nullptr, BS, 2 * HDc, Dc,
        0L, (long)Dc * 2 * HDc, (long)BS * 2 * HDc);
    gemm_kernel<1, 0><<<gq, blk, 0, stream>>>(
        xb, Wk + woff, kb, nullptr, BS, 2 * HDc, Dc,
        0L, (long)Dc * 2 * HDc, (long)BS * 2 * HDc);
    gemm_kernel<1, 0><<<gq, blk, 0, stream>>>(
        xb, Wv + woff, vb, nullptr, BS, 2 * HDc, Dc,
        0L, (long)Dc * 2 * HDc, (long)BS * 2 * HDc);

    // Two flash softmax passes (component = blockIdx.z & 1)
    const dim3 gf(Sc / 128, Hc, Bc * 2);
    flash_kernel<<<gf, blk, 0, stream>>>(qb, kb, vb, o1, o2, scale, Sc, BS);

    combine_kernel<<<BS, blk, 0, stream>>>(o1, o2, lam, ob, BS,
                                           1.f - LAMBDA_INIT);

    // Output projection + residual
    const dim3 go(Dc / 128, BS / 128, 1);
    gemm_kernel<0, 1><<<go, blk, 0, stream>>>(
        ob, Wo + (long)l * 2 * Dc * Dc, yb, xcur, BS, Dc, 2 * Dc, 0L, 0L, 0L);

    // Shared SwiGLU FFN + residual
    rmsnorm_kernel<<<BS, blk, 0, stream>>>(yb, hb, Dc);
    const dim3 g13(HFFc / 128, BS / 128, 1);
    gemm_kernel<0, 0><<<g13, blk, 0, stream>>>(hb, w1, g1, nullptr, BS, HFFc,
                                               Dc, 0L, 0L, 0L);
    gemm_kernel<0, 0><<<g13, blk, 0, stream>>>(hb, w3, g3, nullptr, BS, HFFc,
                                               Dc, 0L, 0L, 0L);
    const long nsw = (long)BS * HFFc;
    swiglu_kernel<<<(unsigned)((nsw + 255) / 256), blk, 0, stream>>>(g1, g3,
                                                                     ub, nsw);
    gemm_kernel<0, 1><<<go, blk, 0, stream>>>(ub, w2, xnext[l], yb, BS, Dc,
                                              HFFc, 0L, 0L, 0L);
    xcur = xnext[l];
  }

  // Vocab projection: out = x @ Wout + bout
  const long ncv = (long)BS * Dc;
  f32_to_bf16_kernel<<<(unsigned)((ncv + 255) / 256), blk, 0, stream>>>(xcur,
                                                                        xb, ncv);
  const dim3 gv(Vc / 128, BS / 128, 1);
  gemm_kernel<0, 2><<<gv, blk, 0, stream>>>(xb, Wout, out, bout, BS, Vc, Dc,
                                            0L, 0L, 0L);
}
